// LanguageModel_38225208934809
// MI455X (gfx1250) — matrix operand order repaired and checked by reading, not compiled, not run on hardware
//
#include <hip/hip_runtime.h>
#include <hip/hip_bf16.h>

// ---------------------------------------------------------------------------
// LSTM language model on MI455X (gfx1250, wave32, WMMA 16x16x32 f16)
//
//   BATCH=16 == WMMA M.  Persistent recurrent kernel: 16 blocks, each owns a
//   64-wide hidden slice (=> 256 gate columns = 16 WMMA N-tiles, 2 per wave).
//   Weights pre-packed f32->f16 into WMMA B-fragment order, streamed from L2
//   (28MB recurrent + 64MB Wout << 192MB L2).  Cell state lives in LDS,
//   h0/h1 double-buffered in global with a device-scope sense barrier.
//   A-operand staging uses CDNA5 async memory->LDS loads (ASYNCcnt).
//   LM head blocks 2 timesteps per workgroup to halve Wout L2 traffic.
// ---------------------------------------------------------------------------

#define D_EMB   512
#define D_RNN   1024
#define VOCAB   32000
#define BATCH   16
#define SEQ     256
#define NBLK    16          // persistent LSTM grid
#define SQRT_DEMB 22.62741699796952f

typedef __attribute__((ext_vector_type(16))) _Float16 v16h;
typedef __attribute__((ext_vector_type(8)))  float    v8f;

// ---------------- workspace layout (bytes) ----------------
#define OFF_W0P   (size_t)0                       // 512*4096 f16   = 4 MiB
#define OFF_U0P   (OFF_W0P  + (size_t)512*4096*2) // 1024*4096 f16  = 8 MiB
#define OFF_W1P   (OFF_U0P  + (size_t)1024*4096*2)
#define OFF_U1P   (OFF_W1P  + (size_t)1024*4096*2)
#define OFF_WOUTP (OFF_U1P  + (size_t)1024*4096*2)   // 1024*32000 f16 = 62.5 MiB
#define OFF_XBUF  (OFF_WOUTP+ (size_t)1024*32000*2)  // [S][B][512] f16
#define OFF_HS    (OFF_XBUF + (size_t)SEQ*BATCH*D_EMB*2) // [S][B][1024] f16
#define OFF_H0    (OFF_HS   + (size_t)SEQ*BATCH*D_RNN*2) // 2x[B][1024] f16
#define OFF_H1    (OFF_H0   + (size_t)2*BATCH*D_RNN*2)
#define OFF_BAR   (OFF_H1   + (size_t)2*BATCH*D_RNN*2)   // 512 u32

// ---------------- helpers ----------------
__device__ __forceinline__ float sigf(float x) {
    return 1.0f / (1.0f + __expf(-x));
}
__device__ __forceinline__ float tanhfast(float x) {
    float e = __expf(2.0f * x);
    return (e - 1.0f) / (e + 1.0f);
}

// CDNA5 async memory->LDS copy (GLOBAL_LOAD_ASYNC_TO_LDS_B128, ASYNCcnt).
// Generic LDS pointers carry the LDS byte offset in their low 32 bits.
__device__ __forceinline__ void async_copy_to_lds(_Float16* dstLds,
                                                  const _Float16* src, int halves) {
    const unsigned ldsbase = (unsigned)(uintptr_t)dstLds;
    const int n = halves >> 3;                    // 16B chunks
    for (int i = threadIdx.x; i < n; i += blockDim.x) {
        unsigned    ldst = ldsbase + (unsigned)i * 16u;
        const char* gsrc = (const char*)src + (size_t)i * 16u;
        asm volatile("global_load_async_to_lds_b128 %0, %1, off"
                     :: "v"(ldst), "v"(gsrc) : "memory");
    }
    asm volatile("s_wait_asynccnt 0x0" ::: "memory");  // own loads complete
}

// Device-scope barrier (NBLK resident blocks), one counter per (step,phase).
__device__ __forceinline__ void grid_barrier(unsigned* bar, int idx) {
    __syncthreads();
    if (threadIdx.x == 0) {
        __builtin_amdgcn_fence(__ATOMIC_RELEASE, "agent");
        __hip_atomic_fetch_add(&bar[idx], 1u, __ATOMIC_RELAXED, __HIP_MEMORY_SCOPE_AGENT);
        while (__hip_atomic_load(&bar[idx], __ATOMIC_RELAXED, __HIP_MEMORY_SCOPE_AGENT)
               < (unsigned)NBLK) {
            __builtin_amdgcn_s_sleep(2);
        }
        __builtin_amdgcn_fence(__ATOMIC_ACQUIRE, "agent");
    }
    __syncthreads();
}

// Accumulate two 16x16 output tiles: C += A(16xK,LDS) x B(Kx32, packed global).
// A fragment (16-bit A 16x32 ISA layout): lane m = l&15, halves 0..7 hold
// K = kb..kb+7, halves 8..15 hold K = kb+16..kb+23, kb = 8*(l>=16).
// B packed as [ntile][KC][lane][16] halves so each lane does one 32B load.
template <int KC>
__device__ __forceinline__ void gemm_acc2(const _Float16* __restrict__ Bp, int nt0,
                                          const _Float16* As, int lane,
                                          v8f& acc0, v8f& acc1) {
    const int m      = lane & 15;
    const int kbase  = (lane & 16) ? 8 : 0;
    const int rowstr = KC * 32 * 2;         // bytes per A row (K = KC*32)
#pragma unroll 4
    for (int kc = 0; kc < KC; ++kc) {
        union { v16h v; uint4 q[2]; } af;
        const char* ap = (const char*)As + m * rowstr + (kc * 32 + kbase) * 2;
        af.q[0] = *(const uint4*)ap;
        af.q[1] = *(const uint4*)(ap + 32);

        const _Float16* bp0 = Bp + ((size_t)(nt0 * KC + kc) * 32 + lane) * 16;
        const _Float16* bp1 = Bp + ((size_t)((nt0 + 1) * KC + kc) * 32 + lane) * 16;
        __builtin_prefetch((const void*)(bp0 + 512), 0, 1);  // next k-chunk
        v16h b0 = *(const v16h*)bp0;
        v16h b1 = *(const v16h*)bp1;
        acc0 = __builtin_amdgcn_wmma_f32_16x16x32_f16(false, af.v, false, b0,
                                                      (short)0, acc0, false, false);
        acc1 = __builtin_amdgcn_wmma_f32_16x16x32_f16(false, af.v, false, b1,
                                                      (short)0, acc1, false, false);
    }
}

// Store one 16x16 f32 accumulator tile into zbuf[16][256] at column lc.
__device__ __forceinline__ void store_ztile(float* zbuf, const v8f& a, int lc, int lane) {
    const int rb = (lane & 16) ? 8 : 0;
#pragma unroll
    for (int r = 0; r < 8; ++r) zbuf[(rb + r) * 256 + lc] = a[r];
}

// Elementwise LSTM gate update for this block's 64-wide hidden slice.
// zbuf layout: [b][gate*64 + jl], gates ordered i,f,g,o.
__device__ __forceinline__ void lstm_gates(const float* zb, float* cs,
                                           const float* __restrict__ bias, int j0,
                                           _Float16* __restrict__ hdst,
                                           _Float16* __restrict__ hdst2) {
#pragma unroll
    for (int ii = 0; ii < 4; ++ii) {
        const int idx = threadIdx.x + ii * 256;   // 0..1023 = b*64 + jl
        const int b   = idx >> 6;
        const int jl  = idx & 63;
        const int jg  = j0 + jl;
        float zi = zb[b * 256 + jl]        + bias[jg];
        float zf = zb[b * 256 + 64 + jl]   + bias[1024 + jg];
        float zg = zb[b * 256 + 128 + jl]  + bias[2048 + jg];
        float zo = zb[b * 256 + 192 + jl]  + bias[3072 + jg];
        float c  = sigf(zf) * cs[idx] + sigf(zi) * tanhfast(zg);
        cs[idx]  = c;
        float h  = sigf(zo) * tanhfast(c);
        hdst[b * D_RNN + jg] = (_Float16)h;
        if (hdst2) hdst2[b * D_RNN + jg] = (_Float16)h;
    }
}

// ---------------- kernels ----------------

__global__ void init_kernel(_Float16* h0g, _Float16* h1g, unsigned* bar) {
    const int t = blockIdx.x * blockDim.x + threadIdx.x;
    const int stride = gridDim.x * blockDim.x;
    for (int i = t; i < 2 * BATCH * D_RNN; i += stride) {
        h0g[i] = (_Float16)0.0f;
        h1g[i] = (_Float16)0.0f;
    }
    for (int i = t; i < 512; i += stride) bar[i] = 0u;
}

// x[s][b][d] = emb[seq[b][s]][d] * sqrt(D_EMB), cast to f16.
__global__ void embed_kernel(const int* __restrict__ seq, const float* __restrict__ emb,
                             _Float16* __restrict__ x) {
    const int b = blockIdx.x;
    const int s = blockIdx.y;
    const int tok = seq[b * SEQ + s];
    const float* e = emb + (size_t)tok * D_EMB;
    _Float16* xo = x + ((size_t)s * BATCH + b) * D_EMB;
    for (int d = threadIdx.x; d < D_EMB; d += blockDim.x)
        xo[d] = (_Float16)(e[d] * SQRT_DEMB);
}

// Pack f32 [K][N] row-major weights into f16 WMMA-B fragment order:
// dst[(nt*KC + kc)*32 + lane][16] with lane = n%16 + 16*((k%32)/8%2), half = k%8 + 8*(k%32>=16).
__global__ void pack_b_f16(const float* __restrict__ src, _Float16* __restrict__ dst,
                           int K, int N) {
    const size_t total = (size_t)K * N;
    const size_t KC = (size_t)(K >> 5);
    for (size_t i = (size_t)blockIdx.x * blockDim.x + threadIdx.x; i < total;
         i += (size_t)gridDim.x * blockDim.x) {
        const int k = (int)(i / N);
        const int n = (int)(i % N);
        const int nt = n >> 4, nn = n & 15;
        const int kc = k >> 5, kk = k & 31;
        const int lane = nn + ((kk & 8) ? 16 : 0);
        const int half = (kk & 7) + ((kk & 16) ? 8 : 0);
        const size_t di = ((((size_t)nt * KC + kc) * 32 + lane) << 4) + half;
        dst[di] = (_Float16)src[i];
    }
}

// Persistent 2-layer LSTM over all SEQ steps.  16 blocks x 256 threads.
__global__ void __launch_bounds__(256)
lstm_persistent(const _Float16* __restrict__ xbuf,
                const _Float16* __restrict__ W0p, const _Float16* __restrict__ U0p,
                const _Float16* __restrict__ W1p, const _Float16* __restrict__ U1p,
                const float* __restrict__ b0, const float* __restrict__ b1,
                _Float16* __restrict__ h0g, _Float16* __restrict__ h1g,
                _Float16* __restrict__ hsg, unsigned* __restrict__ bar) {
    __shared__ _Float16 Abuf[BATCH * D_RNN];   // 32 KB, reused per GEMM pass
    __shared__ float    zbuf[BATCH * 256];     // 16 KB gate pre-activations
    __shared__ float    c0s[BATCH * 64];       // cell state, this block's slice
    __shared__ float    c1s[BATCH * 64];

    const int tid  = threadIdx.x;
    const int lane = tid & 31;
    const int wave = tid >> 5;                 // 8 waves
    const int g    = wave >> 1;                // gate 0..3 (i,f,g,o)
    const int half = wave & 1;                 // which 32-col half of slice
    const int j0   = blockIdx.x * 64;          // hidden slice base
    const int lc0  = g * 64 + half * 32 + (lane & 15);
    const int nt0  = g * 64 + (j0 >> 4) + half * 2;  // N-tile base in [0,256)

    for (int i = tid; i < BATCH * 64; i += 256) { c0s[i] = 0.0f; c1s[i] = 0.0f; }
    __syncthreads();

    const v8f vz = {0.f, 0.f, 0.f, 0.f, 0.f, 0.f, 0.f, 0.f};

    for (int s = 0; s < SEQ; ++s) {
        const int wp = s & 1, rp = wp ^ 1;

        // ================= layer 0: z = x_t @ W0 + h0_prev @ U0 =============
        v8f acc0 = vz, acc1 = vz;
        async_copy_to_lds(Abuf, xbuf + (size_t)s * BATCH * D_EMB, BATCH * D_EMB);
        __syncthreads();
        gemm_acc2<16>(W0p, nt0, Abuf, lane, acc0, acc1);
        __syncthreads();
        async_copy_to_lds(Abuf, h0g + rp * BATCH * D_RNN, BATCH * D_RNN);
        __syncthreads();
        gemm_acc2<32>(U0p, nt0, Abuf, lane, acc0, acc1);
        store_ztile(zbuf, acc0, lc0, lane);
        store_ztile(zbuf, acc1, lc0 + 16, lane);
        __syncthreads();
        lstm_gates(zbuf, c0s, b0, j0, h0g + wp * BATCH * D_RNN, nullptr);
        grid_barrier(bar, s * 2);              // full h0(s) visible device-wide

        // ================= layer 1: z = h0 @ W1 + h1_prev @ U1 ==============
        acc0 = vz; acc1 = vz;
        async_copy_to_lds(Abuf, h0g + wp * BATCH * D_RNN, BATCH * D_RNN);
        __syncthreads();
        gemm_acc2<32>(W1p, nt0, Abuf, lane, acc0, acc1);
        __syncthreads();
        async_copy_to_lds(Abuf, h1g + rp * BATCH * D_RNN, BATCH * D_RNN);
        __syncthreads();
        gemm_acc2<32>(U1p, nt0, Abuf, lane, acc0, acc1);
        store_ztile(zbuf, acc0, lc0, lane);
        store_ztile(zbuf, acc1, lc0 + 16, lane);
        __syncthreads();
        lstm_gates(zbuf, c1s, b1, j0, h1g + wp * BATCH * D_RNN,
                   hsg + (size_t)s * BATCH * D_RNN);
        grid_barrier(bar, s * 2 + 1);          // h1(s) + hs(s) visible
    }
}

// logits[b][s][v] = hs[s][b][:] @ Wout[:,v] + bout[v].
// Grid: (50 n-blocks, 128 s-pairs).  Each block: 8 waves x 5 N-tiles = 640
// cols, TWO timesteps (two A tiles in LDS) per B-fragment load -> halves
// Wout L2 traffic and doubles WMMA work per loaded B byte.
__global__ void __launch_bounds__(256)
lm_head(const _Float16* __restrict__ hsg, const _Float16* __restrict__ Wp,
        const float* __restrict__ bout, float* __restrict__ out) {
    __shared__ _Float16 Abuf0[BATCH * D_RNN];  // hs[s0]: 16x1024 f16 (32 KB)
    __shared__ _Float16 Abuf1[BATCH * D_RNN];  // hs[s1]: 16x1024 f16 (32 KB)
    const int s0 = blockIdx.y * 2;
    const int s1 = s0 + 1;
    async_copy_to_lds(Abuf0, hsg + (size_t)s0 * BATCH * D_RNN, BATCH * D_RNN);
    async_copy_to_lds(Abuf1, hsg + (size_t)s1 * BATCH * D_RNN, BATCH * D_RNN);
    __syncthreads();

    const int lane  = threadIdx.x & 31;
    const int wave  = threadIdx.x >> 5;
    const int ntb   = blockIdx.x * 40 + wave * 5;  // N-tile base in [0,2000)
    const int m     = lane & 15;
    const int kbase = (lane & 16) ? 8 : 0;

    const v8f vz = {0.f, 0.f, 0.f, 0.f, 0.f, 0.f, 0.f, 0.f};
    v8f accA[5], accB[5];
#pragma unroll
    for (int j = 0; j < 5; ++j) { accA[j] = vz; accB[j] = vz; }

#pragma unroll 2
    for (int kc = 0; kc < 32; ++kc) {
        union { v16h v; uint4 q[2]; } af0, af1;
        const int abyte = m * (D_RNN * 2) + (kc * 32 + kbase) * 2;
        af0.q[0] = *(const uint4*)((const char*)Abuf0 + abyte);
        af0.q[1] = *(const uint4*)((const char*)Abuf0 + abyte + 32);
        af1.q[0] = *(const uint4*)((const char*)Abuf1 + abyte);
        af1.q[1] = *(const uint4*)((const char*)Abuf1 + abyte + 32);
#pragma unroll
        for (int j = 0; j < 5; ++j) {
            const _Float16* bp = Wp + ((size_t)(ntb + j) * 32 + kc) * 512 + lane * 16;
            __builtin_prefetch((const void*)(bp + 512), 0, 1);
            v16h b = *(const v16h*)bp;
            accA[j] = __builtin_amdgcn_wmma_f32_16x16x32_f16(false, af0.v, false, b,
                                                             (short)0, accA[j], false, false);
            accB[j] = __builtin_amdgcn_wmma_f32_16x16x32_f16(false, af1.v, false, b,
                                                             (short)0, accB[j], false, false);
        }
    }

    const int rb = (lane & 16) ? 8 : 0;
#pragma unroll
    for (int j = 0; j < 5; ++j) {
        const int v0 = (ntb + j) * 16 + m;
        const float bias = bout[v0];
#pragma unroll
        for (int r = 0; r < 8; ++r) {
            const int bb = rb + r;                        // batch row
            out[((size_t)bb * SEQ + s0) * VOCAB + v0] = accA[j][r] + bias;
            out[((size_t)bb * SEQ + s1) * VOCAB + v0] = accB[j][r] + bias;
        }
    }
}

// ---------------- host entry ----------------
extern "C" void kernel_launch(void* const* d_in, const int* in_sizes, int n_in,
                              void* d_out, int out_size, void* d_ws, size_t ws_size,
                              hipStream_t stream) {
    const int*   seq  = (const int*)  d_in[0];
    const float* emb  = (const float*)d_in[1];
    const float* W0   = (const float*)d_in[2];
    const float* U0   = (const float*)d_in[3];
    const float* b0   = (const float*)d_in[4];
    const float* W1   = (const float*)d_in[5];
    const float* U1   = (const float*)d_in[6];
    const float* b1   = (const float*)d_in[7];
    const float* Wout = (const float*)d_in[8];
    const float* bout = (const float*)d_in[9];
    float* out = (float*)d_out;

    char* ws = (char*)d_ws;
    _Float16* W0p   = (_Float16*)(ws + OFF_W0P);
    _Float16* U0p   = (_Float16*)(ws + OFF_U0P);
    _Float16* W1p   = (_Float16*)(ws + OFF_W1P);
    _Float16* U1p   = (_Float16*)(ws + OFF_U1P);
    _Float16* Woutp = (_Float16*)(ws + OFF_WOUTP);
    _Float16* xbuf  = (_Float16*)(ws + OFF_XBUF);
    _Float16* hsg   = (_Float16*)(ws + OFF_HS);
    _Float16* h0g   = (_Float16*)(ws + OFF_H0);
    _Float16* h1g   = (_Float16*)(ws + OFF_H1);
    unsigned* bar   = (unsigned*)(ws + OFF_BAR);

    init_kernel<<<32, 256, 0, stream>>>(h0g, h1g, bar);
    embed_kernel<<<dim3(BATCH, SEQ), 256, 0, stream>>>(seq, emb, xbuf);

    pack_b_f16<<<8192,   256, 0, stream>>>(W0,   W0p,   D_EMB, 4 * D_RNN);
    pack_b_f16<<<16384,  256, 0, stream>>>(U0,   U0p,   D_RNN, 4 * D_RNN);
    pack_b_f16<<<16384,  256, 0, stream>>>(W1,   W1p,   D_RNN, 4 * D_RNN);
    pack_b_f16<<<16384,  256, 0, stream>>>(U1,   U1p,   D_RNN, 4 * D_RNN);
    pack_b_f16<<<128000, 256, 0, stream>>>(Wout, Woutp, D_RNN, VOCAB);

    lstm_persistent<<<NBLK, 256, 0, stream>>>(xbuf, W0p, U0p, W1p, U1p,
                                              b0, b1, h0g, h1g, hsg, bar);

    lm_head<<<dim3(VOCAB / 640, SEQ / 2), 256, 0, stream>>>(hsg, Woutp, bout, out);
}
